// HeteroAttLayer_36550171689027
// MI455X (gfx1250) — matrix operand order repaired and checked by reading, not compiled, not run on hardware
//
#include <hip/hip_runtime.h>
#include <hip/hip_bf16.h>

// ---------------------------------------------------------------------------
// HeteroAttLayer on MI455X (gfx1250, wave32, WMMA bf16 path)
//   pass 1: scores[c] = sum_{n,d} tanh(hs[c,n,:] @ W^T + b)[d] * meta[c,d]
//   pass 2: out[n,:]  = sum_c softmax(scores/N)[c] * tanh(hs[c,n,:] @ W^T + b)
// v_wmma_f32_16x16x32_bf16, fp32 accumulate. W staged in LDS as packed bf16x2.
// Wave tile = 16 rows x 64 cols (acc 32 VGPR + outAcc 32 VGPR) to stay well
// under 256 VGPRs -> no s_set_vgpr_msb, better occupancy for HBM streaming.
// ---------------------------------------------------------------------------

constexpr int DDIM  = 128;  // att_dim
constexpr int LDSPU = 72;   // uints per W row in LDS: 288B pitch (32B aligned,
                            // 72 banks stride -> staggered, conflict-free-ish)

typedef __attribute__((ext_vector_type(16))) __bf16 v16bf;
typedef __attribute__((ext_vector_type(8)))  float  v8f;

struct alignas(16) F4 { float x, y, z, w; };
struct alignas(16) U4 { unsigned x, y, z, w; };

union BFrag {
  v16bf    v;
  unsigned u[8];
  U4       q[2];
};

// two f32 -> packed bf16x2 (round-half-up + v_perm_b32 byte pack: 3 VALU / 2 elems)
__device__ __forceinline__ unsigned pk2(float lo, float hi) {
  union { float f; unsigned u; } a, b;
  a.f = lo; b.f = hi;
  unsigned au = a.u + 0x8000u;
  unsigned bu = b.u + 0x8000u;
  return __builtin_amdgcn_perm(bu, au, 0x07060302u); // {hi16(bu), hi16(au)}
}

// fast tanh via hardware v_exp_f32 / v_rcp_f32 (~5 VALU, |err| ~1e-6)
__device__ __forceinline__ float fast_tanh(float x) {
  float e = __builtin_amdgcn_exp2f(x * 2.885390082f); // exp(2x)
  float r = __builtin_amdgcn_rcpf(e + 1.0f);
  return 1.0f - 2.0f * r;
}

// Stage W (row-major [e][d], e = output column, d = K) into LDS, bf16x2-packed.
__device__ __forceinline__ void stage_W(const float* __restrict__ W, unsigned* sW) {
  for (int i = threadIdx.x; i < DDIM * (DDIM / 2); i += blockDim.x) {
    int e = i >> 6, dp = i & 63;
    const float* wp = W + e * DDIM + dp * 2;
    sW[e * LDSPU + dp] = pk2(wp[0], wp[1]);
  }
}

// A fragment (16x32 bf16, MxK) per ISA layout:
//   lanes 0-15 : M=lane,    K = kb+{0..7, 16..23}
//   lanes 16-31: M=lane-16, K = kb+{8..15, 24..31}
__device__ __forceinline__ v16bf load_A(const float* __restrict__ rowp, int kbase, int lh) {
  const F4* p = (const F4*)(rowp + kbase + lh * 8);
  F4 f0 = p[0], f1 = p[1], f2 = p[4], f3 = p[5];
  BFrag a;
  a.u[0] = pk2(f0.x, f0.y); a.u[1] = pk2(f0.z, f0.w);
  a.u[2] = pk2(f1.x, f1.y); a.u[3] = pk2(f1.z, f1.w);
  a.u[4] = pk2(f2.x, f2.y); a.u[5] = pk2(f2.z, f2.w);
  a.u[6] = pk2(f3.x, f3.y); a.u[7] = pk2(f3.z, f3.w);
  return a.v;
}

// B fragment (32x16 bf16, KxN): lane = N column, low half K {0..7,16..23}, high half
// K {8..15,24..31}. Two 16B chunks in the LDS W row -> two ds_load_b128.
__device__ __forceinline__ v16bf load_B(const unsigned* sW, int col, int kbase, int lh) {
  const unsigned* wp = sW + col * LDSPU + (kbase >> 1) + lh * 4;
  BFrag b;
  b.q[0] = *(const U4*)(wp);
  b.q[1] = *(const U4*)(wp + 8);
  return b.v;
}

__global__ void __launch_bounds__(256) zero_ws_kernel(float* ws, int n) {
  int i = blockIdx.x * blockDim.x + threadIdx.x;
  if (i < n) ws[i] = 0.0f;
}

// Block = 256 thr = 8 waves = 4 row-tiles x 2 col-halves -> 64 rows per block.
__global__ void __launch_bounds__(256)
score_kernel(const float* __restrict__ hs, const float* __restrict__ W,
             const float* __restrict__ bias, const float* __restrict__ meta,
             float* __restrict__ scores, int N, int C) {
  __shared__ unsigned sW[DDIM * LDSPU];
  __shared__ float    s_score[8];
  stage_W(W, sW);
  if (threadIdx.x < 8) s_score[threadIdx.x] = 0.0f;
  __syncthreads();

  const int lane = threadIdx.x & 31;
  const int wave = threadIdx.x >> 5;
  const int rt   = wave >> 1;          // row tile 0..3
  const int chf  = wave & 1;           // column half 0..1
  const int lh   = lane >> 4;
  const int ln   = lane & 15;
  const int rbase = blockIdx.x * 64 + rt * 16;
  int arow = rbase + ln;
  if (arow > N - 1) arow = N - 1;

  float bcol[4];
#pragma unroll
  for (int t = 0; t < 4; ++t) bcol[t] = bias[chf * 64 + t * 16 + ln];

  for (int c = 0; c < C; ++c) {
    const float* rowp = hs + ((size_t)c * N + arow) * DDIM;
    v8f z = {};
    v8f acc[4];
#pragma unroll
    for (int t = 0; t < 4; ++t) acc[t] = z;

#pragma unroll
    for (int kb = 0; kb < 4; ++kb) {
      v16bf a = load_A(rowp, kb * 32, lh);
#pragma unroll
      for (int t = 0; t < 4; ++t) {
        v16bf bf = load_B(sW, chf * 64 + t * 16 + ln, kb * 32, lh);
        acc[t] = __builtin_amdgcn_wmma_f32_16x16x32_bf16(
            false, a, false, bf, (short)0, acc[t], false, false);
      }
    }

    float part = 0.0f;
#pragma unroll
    for (int t = 0; t < 4; ++t) {
      float mv = meta[c * DDIM + chf * 64 + t * 16 + ln];
#pragma unroll
      for (int j = 0; j < 8; ++j) {
        int row = rbase + lh * 8 + j;               // C/D layout: M = j + 8*half
        float hval = fast_tanh(acc[t][j] + bcol[t]);
        part += (row < N) ? hval * mv : 0.0f;
      }
    }
#pragma unroll
    for (int s = 16; s > 0; s >>= 1) part += __shfl_xor(part, s, 32);
    if (lane == 0) atomicAdd(&s_score[c], part);
  }
  __syncthreads();
  if (threadIdx.x < C) atomicAdd(&scores[threadIdx.x], s_score[threadIdx.x]);
}

__global__ void softmax_kernel(float* __restrict__ ws, int N, int C) {
  // ws[0..C) = raw score sums; ws[C..2C) = softmax(score/N)
  if (threadIdx.x == 0) {
    float s[8];
    float m = -3.402823466e38f;
    for (int c = 0; c < C; ++c) { s[c] = ws[c] / (float)N; m = fmaxf(m, s[c]); }
    float sum = 0.0f;
    for (int c = 0; c < C; ++c) { s[c] = expf(s[c] - m); sum += s[c]; }
    for (int c = 0; c < C; ++c) ws[C + c] = s[c] / sum;
  }
}

__global__ void __launch_bounds__(256)
aggregate_kernel(const float* __restrict__ hs, const float* __restrict__ W,
                 const float* __restrict__ bias, const float* __restrict__ att,
                 float* __restrict__ out, int N, int C) {
  __shared__ unsigned sW[DDIM * LDSPU];
  stage_W(W, sW);
  __syncthreads();

  const int lane = threadIdx.x & 31;
  const int wave = threadIdx.x >> 5;
  const int rt   = wave >> 1;
  const int chf  = wave & 1;
  const int lh   = lane >> 4;
  const int ln   = lane & 15;
  const int rbase = blockIdx.x * 64 + rt * 16;
  int arow = rbase + ln;
  if (arow > N - 1) arow = N - 1;

  float bcol[4];
#pragma unroll
  for (int t = 0; t < 4; ++t) bcol[t] = bias[chf * 64 + t * 16 + ln];

  v8f z = {};
  v8f outAcc[4];
#pragma unroll
  for (int t = 0; t < 4; ++t) outAcc[t] = z;

  for (int c = 0; c < C; ++c) {
    const float* rowp = hs + ((size_t)c * N + arow) * DDIM;
    float ac = att[c];
    v8f acc[4];
#pragma unroll
    for (int t = 0; t < 4; ++t) acc[t] = z;

#pragma unroll
    for (int kb = 0; kb < 4; ++kb) {
      v16bf a = load_A(rowp, kb * 32, lh);
#pragma unroll
      for (int t = 0; t < 4; ++t) {
        v16bf bf = load_B(sW, chf * 64 + t * 16 + ln, kb * 32, lh);
        acc[t] = __builtin_amdgcn_wmma_f32_16x16x32_bf16(
            false, a, false, bf, (short)0, acc[t], false, false);
      }
    }
#pragma unroll
    for (int t = 0; t < 4; ++t)
#pragma unroll
      for (int j = 0; j < 8; ++j)
        outAcc[t][j] += ac * fast_tanh(acc[t][j] + bcol[t]);
  }

#pragma unroll
  for (int t = 0; t < 4; ++t) {
    int col = chf * 64 + t * 16 + ln;
#pragma unroll
    for (int j = 0; j < 8; ++j) {
      int row = rbase + lh * 8 + j;
      if (row < N) out[(size_t)row * DDIM + col] = outAcc[t][j];
    }
  }
}

extern "C" void kernel_launch(void* const* d_in, const int* in_sizes, int n_in,
                              void* d_out, int out_size, void* d_ws, size_t ws_size,
                              hipStream_t stream) {
  const float* hs   = (const float*)d_in[0];
  const float* W    = (const float*)d_in[1];
  const float* bias = (const float*)d_in[2];
  const float* meta = (const float*)d_in[3];
  const int C = in_sizes[3] / DDIM;           // 4
  const int N = in_sizes[0] / (C * DDIM);     // 300000
  float* ws  = (float*)d_ws;                  // [0..C) scores, [C..2C) att
  float* out = (float*)d_out;

  zero_ws_kernel<<<1, 256, 0, stream>>>(ws, 2 * C);

  const int blocks = (N + 63) / 64;
  score_kernel<<<blocks, 256, 0, stream>>>(hs, W, bias, meta, ws, N, C);
  softmax_kernel<<<1, 32, 0, stream>>>(ws, N, C);
  aggregate_kernel<<<blocks, 256, 0, stream>>>(hs, W, bias, ws + C, out, N, C);
}
